// TransformerSelfAttention_3573412790818
// MI455X (gfx1250) — matrix operand order repaired and checked by reading, not compiled, not run on hardware
//
#include <hip/hip_runtime.h>
#include <hip/hip_bf16.h>

typedef __attribute__((ext_vector_type(16))) _Float16 v16h;
typedef __attribute__((ext_vector_type(8)))  _Float16 v8h;
typedef __attribute__((ext_vector_type(8)))  float    v8f;

#define NH   16
#define SS   2048
#define DH   64
#define HID  1024
#define BB   2

// ---------------------------------------------------------------------------
// WMMA wave-level primitives (layouts per CDNA5 ISA 7.12.2, wave32)
// ---------------------------------------------------------------------------

static __device__ __forceinline__ v8f wmma_f16(v16h a, v16h b, v8f c) {
  return __builtin_amdgcn_wmma_f32_16x16x32_f16(false, a, false, b, (short)0, c,
                                                false, false);
}

// A-matrix 16x32 f16: lane<16 -> row m, K = k0+[0..7] and k0+[16..23]
//                     lane>=16 -> same row, K = k0+[8..15] and k0+[24..31]
static __device__ __forceinline__ v16h loadA16(const _Float16* __restrict__ base,
                                               int row0, int ld, int k0, int lane) {
  const int m = lane & 15, hf = lane >> 4;
  const _Float16* p = base + (size_t)(row0 + m) * ld + k0 + hf * 8;
  v8h lo = *(const v8h*)(p);
  v8h hi = *(const v8h*)(p + 16);
  return __builtin_shufflevector(lo, hi, 0,1,2,3,4,5,6,7,8,9,10,11,12,13,14,15);
}

// Same A fragment but source is fp32 (convert on the fly)
static __device__ __forceinline__ v16h loadA32(const float* __restrict__ base,
                                               int row0, int ld, int k0, int lane) {
  const int m = lane & 15, hf = lane >> 4;
  const float* p = base + (size_t)(row0 + m) * ld + k0 + hf * 8;
  v16h r;
#pragma unroll
  for (int i = 0; i < 8; ++i) {
    r[i]     = (_Float16)p[i];
    r[8 + i] = (_Float16)p[16 + i];
  }
  return r;
}

// B-matrix 32x16 f16: lane<16 -> col n, K = k0+[0..7] and k0+[16..23]; lane>=16 -> K = k0+[8..15] and k0+[24..31].
// Stored as rows of the "key" matrix (col n == key row n), contiguous along K.
static __device__ __forceinline__ v16h loadB16(const _Float16* __restrict__ base,
                                               int col0, int ld, int k0, int lane) {
  const int n = lane & 15, hf = lane >> 4;
  const _Float16* p = base + (size_t)(col0 + n) * ld + k0 + hf * 8;
  v8h lo = *(const v8h*)(p);
  v8h hi = *(const v8h*)(p + 16);
  return __builtin_shufflevector(lo, hi, 0,1,2,3,4,5,6,7,8,9,10,11,12,13,14,15);
}

// ---------------------------------------------------------------------------
// Kernels
// ---------------------------------------------------------------------------

__global__ void cast_f32_f16(const float* __restrict__ in,
                             _Float16* __restrict__ out, int n) {
  int i = blockIdx.x * blockDim.x + threadIdx.x;
  if (i < n) out[i] = (_Float16)in[i];
}

// Y = X[4096,1024] * W^T[1024,1024] + b  ->  f16, head-split or per-head-transposed.
// 16x64 register tile per wave: one A fragment reused over 4 B fragments, so
// each wave stays within a single head (64 = head_dim).
// mode 0: out[((b*16+h)*2048+s)*64+d]   (Q/K/P)
// mode 1: out[((b*16+h)*64+d)*2048+s]   (V transposed for the PV GEMM)
__global__ __launch_bounds__(256) void proj_gemm(const _Float16* __restrict__ X,
                                                 const _Float16* __restrict__ W,
                                                 const float* __restrict__ bias,
                                                 _Float16* __restrict__ out,
                                                 int mode) {
  const int lane = threadIdx.x & 31;
  const int wave = threadIdx.x >> 5;
  const int tile = blockIdx.x * 8 + wave;   // 256 row-tiles x 16 col-groups
  const int tn = tile & 15;
  const int tm = tile >> 4;
  const int r0 = tm * 16, o0 = tn * 64;

  v8f acc[4] = {};
  for (int k0 = 0; k0 < HID; k0 += 32) {
    v16h a  = loadA16(X, r0, HID, k0, lane);
    v16h b0 = loadB16(W, o0,      HID, k0, lane);
    v16h b1 = loadB16(W, o0 + 16, HID, k0, lane);
    v16h b2 = loadB16(W, o0 + 32, HID, k0, lane);
    v16h b3 = loadB16(W, o0 + 48, HID, k0, lane);
    acc[0] = wmma_f16(a, b0, acc[0]);
    acc[1] = wmma_f16(a, b1, acc[1]);
    acc[2] = wmma_f16(a, b2, acc[2]);
    acc[3] = wmma_f16(a, b3, acc[3]);
  }
  const int n = lane & 15, hf = lane >> 4;
  const int h = o0 >> 6;                    // whole wave inside one head
#pragma unroll
  for (int j = 0; j < 4; ++j) {
    const int o = o0 + j * 16 + n;
    const int d = o & 63;
    const float bvj = bias[o];
#pragma unroll
    for (int vg = 0; vg < 8; ++vg) {
      int r = r0 + vg + 8 * hf;
      int bb = r >> 11, s = r & (SS - 1);
      float y = acc[j][vg] + bvj;
      if (mode == 0)
        out[((size_t)(bb * NH + h) * SS + s) * DH + d] = (_Float16)y;
      else
        out[((size_t)(bb * NH + h) * DH + d) * SS + s] = (_Float16)y;
    }
  }
}

// scores = (Q K^T + Q P^T + P K^T)/8 + mask, fused row softmax.
// One block per (bh, 16-row q block); 8 waves x 16 col-tiles = full 2048 cols
// kept in registers (16 v8f accumulators / wave).
__global__ __launch_bounds__(256) void scores_softmax(
    const _Float16* __restrict__ Qh, const _Float16* __restrict__ Kh,
    const _Float16* __restrict__ Ph, const float* __restrict__ mask,
    float* __restrict__ wout) {
  __shared__ float red[8][16];
  const int lane = threadIdx.x & 31;
  const int wave = threadIdx.x >> 5;
  const int qb = blockIdx.x & 127;
  const int bh = blockIdx.x >> 7;
  const int q0 = qb * 16;
  const int n = lane & 15, hf = lane >> 4;

  const _Float16* Qb = Qh + (size_t)bh * SS * DH;
  const _Float16* Kb = Kh + (size_t)bh * SS * DH;
  const _Float16* Pb = Ph + (size_t)bh * SS * DH;

  v16h qa0 = loadA16(Qb, q0, DH, 0, lane);
  v16h qa1 = loadA16(Qb, q0, DH, 32, lane);
  v16h pa0 = loadA16(Pb, q0, DH, 0, lane);
  v16h pa1 = loadA16(Pb, q0, DH, 32, lane);

  float sc[16][8];
#pragma unroll
  for (int tl = 0; tl < 16; ++tl) {
    const int c0 = (wave + tl * 8) * 16;
    v16h kb0 = loadB16(Kb, c0, DH, 0, lane);
    v16h kb1 = loadB16(Kb, c0, DH, 32, lane);
    v16h pb0 = loadB16(Pb, c0, DH, 0, lane);
    v16h pb1 = loadB16(Pb, c0, DH, 32, lane);
    v8f acc = {};
    acc = wmma_f16(qa0, kb0, acc);   // c2c
    acc = wmma_f16(qa1, kb1, acc);
    acc = wmma_f16(qa0, pb0, acc);   // c2p
    acc = wmma_f16(qa1, pb1, acc);
    acc = wmma_f16(pa0, kb0, acc);   // p2c
    acc = wmma_f16(pa1, kb1, acc);
#pragma unroll
    for (int vg = 0; vg < 8; ++vg) {
      int r = q0 + vg + 8 * hf;
      sc[tl][vg] = acc[vg] * 0.125f + mask[(size_t)r * SS + c0 + n];
    }
  }

  // ---- row max (half-wave shuffle + cross-wave LDS) ----
  float mx[8];
#pragma unroll
  for (int vg = 0; vg < 8; ++vg) {
    float v = -1e30f;
#pragma unroll
    for (int tl = 0; tl < 16; ++tl) v = fmaxf(v, sc[tl][vg]);
    mx[vg] = v;
  }
#pragma unroll
  for (int msk = 1; msk <= 8; msk <<= 1)
#pragma unroll
    for (int vg = 0; vg < 8; ++vg)
      mx[vg] = fmaxf(mx[vg], __shfl_xor(mx[vg], msk, 32));
  if ((lane & 15) == 0)
#pragma unroll
    for (int vg = 0; vg < 8; ++vg) red[wave][hf * 8 + vg] = mx[vg];
  __syncthreads();
#pragma unroll
  for (int vg = 0; vg < 8; ++vg) {
    float v = red[0][hf * 8 + vg];
#pragma unroll
    for (int w = 1; w < 8; ++w) v = fmaxf(v, red[w][hf * 8 + vg]);
    mx[vg] = v;
  }
  __syncthreads();

  // ---- exp + row sum ----
  float sm[8];
#pragma unroll
  for (int vg = 0; vg < 8; ++vg) sm[vg] = 0.0f;
#pragma unroll
  for (int tl = 0; tl < 16; ++tl)
#pragma unroll
    for (int vg = 0; vg < 8; ++vg) {
      float e = __expf(sc[tl][vg] - mx[vg]);
      sc[tl][vg] = e;
      sm[vg] += e;
    }
#pragma unroll
  for (int msk = 1; msk <= 8; msk <<= 1)
#pragma unroll
    for (int vg = 0; vg < 8; ++vg) sm[vg] += __shfl_xor(sm[vg], msk, 32);
  if ((lane & 15) == 0)
#pragma unroll
    for (int vg = 0; vg < 8; ++vg) red[wave][hf * 8 + vg] = sm[vg];
  __syncthreads();
#pragma unroll
  for (int vg = 0; vg < 8; ++vg) {
    float v = red[0][hf * 8 + vg];
#pragma unroll
    for (int w = 1; w < 8; ++w) v += red[w][hf * 8 + vg];
    sm[vg] = 1.0f / v;
  }

  // ---- write probabilities (fp32, straight into d_out weights region) ----
#pragma unroll
  for (int tl = 0; tl < 16; ++tl) {
    const int c0 = (wave + tl * 8) * 16;
#pragma unroll
    for (int vg = 0; vg < 8; ++vg) {
      int r = q0 + vg + 8 * hf;
      wout[((size_t)bh * SS + r) * SS + c0 + n] = sc[tl][vg] * sm[vg];
    }
  }
}

// out_h = weights(f32, from d_out) @ V ; 16x64 tile per wave so the 537MB
// weights tensor is read from HBM exactly once (A frag reused over 4 d-tiles).
__global__ __launch_bounds__(256) void pv_gemm(const float* __restrict__ Wt,
                                               const _Float16* __restrict__ Vt,
                                               _Float16* __restrict__ AO) {
  const int lane = threadIdx.x & 31;
  const int wave = threadIdx.x >> 5;
  const int widx = blockIdx.x * 8 + wave;   // 32 bh * 128 qt
  const int qt = widx & 127;
  const int bh = widx >> 7;
  const int q0 = qt * 16;
  const float* Wb = Wt + (size_t)bh * SS * SS;
  const _Float16* Vb = Vt + (size_t)bh * DH * SS;

  v8f acc[4] = {};
  for (int k0 = 0; k0 < SS; k0 += 32) {
    v16h a  = loadA32(Wb, q0, SS, k0, lane);
    v16h b0 = loadB16(Vb, 0,  SS, k0, lane);
    v16h b1 = loadB16(Vb, 16, SS, k0, lane);
    v16h b2 = loadB16(Vb, 32, SS, k0, lane);
    v16h b3 = loadB16(Vb, 48, SS, k0, lane);
    acc[0] = wmma_f16(a, b0, acc[0]);
    acc[1] = wmma_f16(a, b1, acc[1]);
    acc[2] = wmma_f16(a, b2, acc[2]);
    acc[3] = wmma_f16(a, b3, acc[3]);
  }
  const int n = lane & 15, hf = lane >> 4;
  const int h = bh & 15, bb = bh >> 4;
#pragma unroll
  for (int j = 0; j < 4; ++j)
#pragma unroll
    for (int vg = 0; vg < 8; ++vg) {
      int s = q0 + vg + 8 * hf;
      AO[((size_t)(bb * SS + s)) * HID + h * DH + j * 16 + n] =
          (_Float16)acc[j][vg];
    }
}

// final: out = AO[4096,1024] @ Wo^T + bo  (fp32 out), 16x64 tile per wave
__global__ __launch_bounds__(256) void out_gemm(const _Float16* __restrict__ X,
                                                const _Float16* __restrict__ W,
                                                const float* __restrict__ bias,
                                                float* __restrict__ out) {
  const int lane = threadIdx.x & 31;
  const int wave = threadIdx.x >> 5;
  const int tile = blockIdx.x * 8 + wave;
  const int tn = tile & 15;
  const int tm = tile >> 4;
  const int r0 = tm * 16, o0 = tn * 64;

  v8f acc[4] = {};
  for (int k0 = 0; k0 < HID; k0 += 32) {
    v16h a  = loadA16(X, r0, HID, k0, lane);
    v16h b0 = loadB16(W, o0,      HID, k0, lane);
    v16h b1 = loadB16(W, o0 + 16, HID, k0, lane);
    v16h b2 = loadB16(W, o0 + 32, HID, k0, lane);
    v16h b3 = loadB16(W, o0 + 48, HID, k0, lane);
    acc[0] = wmma_f16(a, b0, acc[0]);
    acc[1] = wmma_f16(a, b1, acc[1]);
    acc[2] = wmma_f16(a, b2, acc[2]);
    acc[3] = wmma_f16(a, b3, acc[3]);
  }
  const int n = lane & 15, hf = lane >> 4;
#pragma unroll
  for (int j = 0; j < 4; ++j) {
    const int o = o0 + j * 16 + n;
    const float bvj = bias[o];
#pragma unroll
    for (int vg = 0; vg < 8; ++vg) {
      int r = r0 + vg + 8 * hf;
      out[(size_t)r * HID + o] = acc[j][vg] + bvj;
    }
  }
}

// ---------------------------------------------------------------------------
// Launch
// ---------------------------------------------------------------------------

extern "C" void kernel_launch(void* const* d_in, const int* in_sizes, int n_in,
                              void* d_out, int out_size, void* d_ws,
                              size_t ws_size, hipStream_t stream) {
  (void)in_sizes; (void)n_in; (void)out_size; (void)ws_size;

  const float* q    = (const float*)d_in[0];
  const float* k    = (const float*)d_in[1];
  const float* v    = (const float*)d_in[2];
  const float* pos  = (const float*)d_in[3];
  const float* mask = (const float*)d_in[4];
  const float* Wq = (const float*)d_in[5];  const float* bq = (const float*)d_in[6];
  const float* Wk = (const float*)d_in[7];  const float* bk = (const float*)d_in[8];
  const float* Wv = (const float*)d_in[9];  const float* bv = (const float*)d_in[10];
  const float* Wp = (const float*)d_in[11]; const float* bp = (const float*)d_in[12];
  const float* Wo = (const float*)d_in[13]; const float* bo = (const float*)d_in[14];

  float* out = (float*)d_out;                      // [B,S,1024] = 4194304 f32
  float* wts = out + (size_t)BB * SS * HID;        // [B,H,S,S] f32 weights

  // ws layout (f16 element offsets)
  _Float16* ws = (_Float16*)d_ws;
  const size_t NX = (size_t)BB * SS * HID;         // 4194304
  const size_t NW = (size_t)HID * HID;             // 1048576
  _Float16* QC = ws;            _Float16* KC = QC + NX;
  _Float16* VC = KC + NX;       _Float16* PC = VC + NX;
  _Float16* WQh = PC + NX;      _Float16* WKh = WQh + NW;
  _Float16* WVh = WKh + NW;     _Float16* WPh = WVh + NW;
  _Float16* WOh = WPh + NW;
  _Float16* QH = WOh + NW;      _Float16* KH = QH + NX;
  _Float16* PH = KH + NX;       _Float16* VT = PH + NX;
  _Float16* AO = VT + NX;

  const int c1 = ((int)NX + 255) / 256;
  const int c2 = ((int)NW + 255) / 256;
  cast_f32_f16<<<c1, 256, 0, stream>>>(q, QC, (int)NX);
  cast_f32_f16<<<c1, 256, 0, stream>>>(k, KC, (int)NX);
  cast_f32_f16<<<c1, 256, 0, stream>>>(v, VC, (int)NX);
  cast_f32_f16<<<c1, 256, 0, stream>>>(pos, PC, (int)NX);
  cast_f32_f16<<<c2, 256, 0, stream>>>(Wq, WQh, (int)NW);
  cast_f32_f16<<<c2, 256, 0, stream>>>(Wk, WKh, (int)NW);
  cast_f32_f16<<<c2, 256, 0, stream>>>(Wv, WVh, (int)NW);
  cast_f32_f16<<<c2, 256, 0, stream>>>(Wp, WPh, (int)NW);
  cast_f32_f16<<<c2, 256, 0, stream>>>(Wo, WOh, (int)NW);

  // projections: 4096 16x64 wave-tiles, 8 waves/block
  proj_gemm<<<512, 256, 0, stream>>>(QC, WQh, bq, QH, 0);
  proj_gemm<<<512, 256, 0, stream>>>(KC, WKh, bk, KH, 0);
  proj_gemm<<<512, 256, 0, stream>>>(PC, WPh, bp, PH, 0);
  proj_gemm<<<512, 256, 0, stream>>>(VC, WVh, bv, VT, 1);

  // fused scores + softmax: one block per (bh, q-block)
  scores_softmax<<<BB * NH * (SS / 16), 256, 0, stream>>>(QH, KH, PH, mask, wts);

  // probabilities @ V (weights read once)
  pv_gemm<<<512, 256, 0, stream>>>(wts, VT, AO);

  // output projection
  out_gemm<<<512, 256, 0, stream>>>(AO, WOh, bo, out);
}
